// MultiHeadAttentionLayer_32117765439855
// MI455X (gfx1250) — matrix operand order repaired and checked by reading, not compiled, not run on hardware
//
#include <hip/hip_runtime.h>
#include <hip/hip_bf16.h>
#include <cstddef>

#define B_  4
#define S_  2048
#define D_  1024
#define H_  16
#define HD_ 64

typedef __attribute__((ext_vector_type(16))) __bf16 v16bf;
typedef __attribute__((ext_vector_type(8)))  __bf16 v8bf;
typedef __attribute__((ext_vector_type(8)))  float  v8f;
typedef __attribute__((ext_vector_type(8)))  unsigned short u16x8;

// ---- CDNA5 async global->LDS copy (guarded; falls back to sync copy) ------
#if defined(__has_builtin)
#if __has_builtin(__builtin_amdgcn_global_load_async_to_lds_b128) && \
    __has_builtin(__builtin_amdgcn_s_wait_asynccnt)
#define USE_ASYNC_LDS 1
#endif
#endif
#ifndef USE_ASYNC_LDS
#define USE_ASYNC_LDS 0
#endif

#if USE_ASYNC_LDS
typedef int vint4 __attribute__((vector_size(16)));   // matches builtin's pointee type
static __device__ __forceinline__ void async_copy16(const void* g, void* l) {
  // generic->AS1 via 64-bit int; generic LDS ptr low 32 bits == LDS offset -> AS3
  __builtin_amdgcn_global_load_async_to_lds_b128(
      (__attribute__((address_space(1))) vint4*)(unsigned long long)g,
      (__attribute__((address_space(3))) vint4*)(unsigned int)(unsigned long long)l,
      0, 0);
}
static __device__ __forceinline__ void async_wait0() {
  __builtin_amdgcn_s_wait_asynccnt(0);
}
#else
static __device__ __forceinline__ void async_copy16(const void* g, void* l) {
  *(u16x8*)l = *(const u16x8*)g;
}
static __device__ __forceinline__ void async_wait0() {}
#endif

static __device__ __forceinline__ unsigned short f2bfu(float f) {
  union { float f; unsigned u; } cv; cv.f = f;
  unsigned u = cv.u;
  u += 0x7FFFu + ((u >> 16) & 1u);   // round-to-nearest-even
  return (unsigned short)(u >> 16);
}

static __device__ __forceinline__ v16bf bfcat(v8bf lo, v8bf hi) {
  return __builtin_shufflevector(lo, hi, 0,1,2,3,4,5,6,7,8,9,10,11,12,13,14,15);
}

static __device__ __forceinline__ v8f wmma_bf16(v16bf a, v16bf b, v8f c) {
  // D = A(16x32 bf16) * B(32x16 bf16) + C(16x16 f32)
  return __builtin_amdgcn_wmma_f32_16x16x32_bf16(false, a, false, b, (short)0, c, false, false);
}

// reductions across the 16 lanes of each wave32 half (offsets < 16 stay in-half)
static __device__ __forceinline__ float half_max(float v) {
  #pragma unroll
  for (int off = 1; off < 16; off <<= 1) v = fmaxf(v, __shfl_xor(v, off, 32));
  return v;
}
static __device__ __forceinline__ float half_sum(float v) {
  #pragma unroll
  for (int off = 1; off < 16; off <<= 1) v += __shfl_xor(v, off, 32);
  return v;
}

// ---------------------------------------------------------------------------
// Kernel 1: W [K=D, N=D] f32 -> Wt [N, K] bf16  (transpose + convert)
// ---------------------------------------------------------------------------
__global__ __launch_bounds__(256)
void transpose_cvt_kernel(const float* __restrict__ W0, const float* __restrict__ W1,
                          const float* __restrict__ W2, const float* __restrict__ W3,
                          unsigned short* __restrict__ T0, unsigned short* __restrict__ T1,
                          unsigned short* __restrict__ T2, unsigned short* __restrict__ T3) {
  __shared__ float tile[32][33];
  const float* W; unsigned short* T;
  switch (blockIdx.z) {
    case 0:  W = W0; T = T0; break;
    case 1:  W = W1; T = T1; break;
    case 2:  W = W2; T = T2; break;
    default: W = W3; T = T3; break;
  }
  int bx = blockIdx.x * 32, by = blockIdx.y * 32;
  int tx = threadIdx.x, ty = threadIdx.y;
  #pragma unroll
  for (int i = 0; i < 4; ++i)
    tile[ty + i*8][tx] = W[(size_t)(by + ty + i*8) * D_ + bx + tx];
  __syncthreads();
  #pragma unroll
  for (int i = 0; i < 4; ++i)
    T[(size_t)(bx + ty + i*8) * D_ + by + tx] = f2bfu(tile[tx][ty + i*8]);
}

// ---------------------------------------------------------------------------
// Kernel 2/5: C = A[8192,1024](f32) @ Wt^T + bias, via bf16 WMMA.
//   Bt is pre-transposed bf16 [N,K].  WG = 128 threads = 4 waves, 64x64 tile,
//   each wave a 32x32 sub-tile (2x2 WMMA accumulators). K-step 32.
// mode 0: out_bf = bf16 in [B,H,S,HD] head layout, value scaled by `scale`
// mode 1: out_f  = f32 row-major [8192,1024]
// ---------------------------------------------------------------------------
__global__ __launch_bounds__(128)
void gemm_proj_kernel(const float* __restrict__ A,
                      const unsigned short* __restrict__ Bt,
                      const float* __restrict__ bias,
                      unsigned short* __restrict__ out_bf,
                      float* __restrict__ out_f,
                      int mode, float scale) {
  __shared__ unsigned short sA[64 * 40];   // 64 rows x 32 K bf16, 80B row stride
  __shared__ unsigned short sB[64 * 40];   // 64 N-rows x 32 K bf16, 80B stride

  const int tid  = threadIdx.x;
  const int lane = tid & 31, wave = tid >> 5;
  const int l16  = lane & 15, g = lane >> 4;
  const int wm   = wave >> 1, wn = wave & 1;
  const int Mt   = blockIdx.y * 64, Nt = blockIdx.x * 64;

  const v8f z8 = {0.f,0.f,0.f,0.f,0.f,0.f,0.f,0.f};
  v8f acc[2][2];
  acc[0][0] = z8; acc[0][1] = z8; acc[1][0] = z8; acc[1][1] = z8;

  for (int kk = 0; kk < D_; kk += 32) {
    __syncthreads();
    // B tile (pure bf16 copy): async global->LDS, 256 16B chunks
    #pragma unroll
    for (int it = 0; it < 2; ++it) {
      int c   = tid + it * 128;
      int row = c >> 2, seg = c & 3;
      async_copy16(Bt + (size_t)(Nt + row) * D_ + kk + seg * 8, &sB[row * 40 + seg * 8]);
    }
    // A tile (f32 -> bf16): batch all loads first so LOADcnt pipelines
    float4 fv[4];
    #pragma unroll
    for (int it = 0; it < 4; ++it) {
      int i   = tid + it * 128;
      int row = i >> 3, c4 = i & 7;
      fv[it] = *(const float4*)(A + (size_t)(Mt + row) * D_ + kk + c4 * 4);
    }
    if (kk + 32 < D_) __builtin_prefetch(A + (size_t)(Mt + (tid >> 1)) * D_ + kk + 32, 0, 1);
    #pragma unroll
    for (int it = 0; it < 4; ++it) {
      int i   = tid + it * 128;
      int row = i >> 3, c4 = i & 7;
      unsigned short* p = &sA[row * 40 + c4 * 4];
      p[0] = f2bfu(fv[it].x); p[1] = f2bfu(fv[it].y);
      p[2] = f2bfu(fv[it].z); p[3] = f2bfu(fv[it].w);
    }
    async_wait0();
    __syncthreads();

    v16bf a[2], b[2];
    #pragma unroll
    for (int ms = 0; ms < 2; ++ms) {  // A operand: lane row = l16, K chunks {8g..}, {16+8g..}
      int r = wm * 32 + ms * 16 + l16;
      a[ms] = bfcat(*(const v8bf*)&sA[r * 40 + g * 8],
                    *(const v8bf*)&sA[r * 40 + 16 + g * 8]);
    }
    #pragma unroll
    for (int ns = 0; ns < 2; ++ns) {  // B operand: lane col = l16, K chunks {8g..}, {16+8g..}
      int r = wn * 32 + ns * 16 + l16;
      b[ns] = bfcat(*(const v8bf*)&sB[r * 40 + g * 8],
                    *(const v8bf*)&sB[r * 40 + 16 + g * 8]);
    }
    #pragma unroll
    for (int ms = 0; ms < 2; ++ms)
      #pragma unroll
      for (int ns = 0; ns < 2; ++ns)
        acc[ms][ns] = wmma_bf16(a[ms], b[ns], acc[ms][ns]);
  }

  // epilogue: C row = r + 8*g, col = l16
  #pragma unroll
  for (int ms = 0; ms < 2; ++ms)
    #pragma unroll
    for (int ns = 0; ns < 2; ++ns) {
      const int col  = Nt + wn * 32 + ns * 16 + l16;
      const float bi = bias[col];
      const int row0 = Mt + wm * 32 + ms * 16 + g * 8;
      #pragma unroll
      for (int r = 0; r < 8; ++r) {
        const int row = row0 + r;
        const float v = (acc[ms][ns][r] + bi) * scale;
        if (mode == 0) {
          const int b_ = row >> 11, s_ = row & (S_ - 1);
          const int h_ = col >> 6,  hd = col & (HD_ - 1);
          out_bf[((size_t)(b_ * H_ + h_) * S_ + s_) * HD_ + hd] = f2bfu(v);
        } else {
          out_f[(size_t)row * D_ + col] = v;
        }
      }
    }
}

// ---------------------------------------------------------------------------
// Kernel 3: flash-style attention per (b, h, 64-query tile).
//  WG = 128 threads = 4 waves; wave w owns query rows [w*16, w*16+16).
//  Pass 1: stream K in 32-key tiles, scores via WMMA, online row max/sum.
//  Pass 2: recompute scores, p = exp(s-m)/l, write attention probs (f32) once,
//          stage p as bf16 in LDS, accumulate O += P@V via WMMA.
// ---------------------------------------------------------------------------
__global__ __launch_bounds__(128)
void attention_kernel(const unsigned short* __restrict__ Qbf,
                      const unsigned short* __restrict__ Kbf,
                      const unsigned short* __restrict__ Vbf,
                      const int* __restrict__ mask,
                      float* __restrict__ att,      // [B,H,S,S]
                      float* __restrict__ Obuf) {   // [B*S, D]
  __shared__ unsigned short sQ[64 * 72];      // 64 q rows x 64 hd, 144B stride
  __shared__ unsigned short sK[32 * 72];      // 32 keys x 64 hd, 144B stride
  __shared__ unsigned short sVt[64 * 40];     // transposed: 64 hd rows x 32 keys, 80B stride
  __shared__ unsigned short sP[4][16 * 40];   // per-wave P staging: 16 q x 32 k, 80B stride

  const int qt = blockIdx.x, h = blockIdx.y, b = blockIdx.z;
  const int tid  = threadIdx.x;
  const int lane = tid & 31, wave = tid >> 5;
  const int l16  = lane & 15, g = lane >> 4;

  const unsigned short* Qb = Qbf + ((size_t)(b * H_ + h) * S_ + qt * 64) * HD_;
  const unsigned short* Kb = Kbf + (size_t)(b * H_ + h) * S_ * HD_;
  const unsigned short* Vb = Vbf + (size_t)(b * H_ + h) * S_ * HD_;
  const int* mrow = mask + (size_t)b * S_;
  float* attb = att + (size_t)(b * H_ + h) * S_ * S_;

  { // load Q tile (64 x 128B): async, thread copies 64B as 4x16B
    const int row = tid >> 1, hf = tid & 1;
    #pragma unroll
    for (int j = 0; j < 4; ++j)
      async_copy16(Qb + row * HD_ + hf * 32 + j * 8, &sQ[row * 72 + hf * 32 + j * 8]);
  }
  async_wait0();
  __syncthreads();

  // resident Q A-operands (hd 0..31 and 32..63)
  v16bf qa0, qa1;
  {
    const unsigned short* p = &sQ[(wave * 16 + l16) * 72];
    qa0 = bfcat(*(const v8bf*)(p + g * 8),      *(const v8bf*)(p + 16 + g * 8));
    qa1 = bfcat(*(const v8bf*)(p + 32 + g * 8), *(const v8bf*)(p + 48 + g * 8));
  }

  float m_[8], l_[8];
  #pragma unroll
  for (int r = 0; r < 8; ++r) { m_[r] = -3.0e38f; l_[r] = 0.f; }

  const v8f z8 = {0.f,0.f,0.f,0.f,0.f,0.f,0.f,0.f};

  // ----------------- pass 1: row max + sum of exp -----------------
  for (int k0 = 0; k0 < S_; k0 += 32) {
    __syncthreads();
    { // K tile: 32 rows x 128B, async, thread copies 32B
      const int row = tid >> 2, seg = tid & 3;
      async_copy16(Kb + (size_t)(k0 + row) * HD_ + seg * 16,     &sK[row * 72 + seg * 16]);
      async_copy16(Kb + (size_t)(k0 + row) * HD_ + seg * 16 + 8, &sK[row * 72 + seg * 16 + 8]);
    }
    if (k0 + 32 < S_) __builtin_prefetch(Kb + (size_t)(k0 + 32 + (tid >> 2)) * HD_, 0, 1);
    async_wait0();
    __syncthreads();

    v8f c[2]; c[0] = z8; c[1] = z8;
    #pragma unroll
    for (int ks = 0; ks < 2; ++ks) {
      const unsigned short* p = &sK[(ks * 16 + l16) * 72];
      v16bf blo = bfcat(*(const v8bf*)(p + g * 8),      *(const v8bf*)(p + 16 + g * 8));
      v16bf bhi = bfcat(*(const v8bf*)(p + 32 + g * 8), *(const v8bf*)(p + 48 + g * 8));
      c[ks] = wmma_bf16(qa0, blo, c[ks]);
      c[ks] = wmma_bf16(qa1, bhi, c[ks]);
    }
    const int mk0 = mrow[k0 + l16];
    const int mk1 = mrow[k0 + 16 + l16];
    #pragma unroll
    for (int r = 0; r < 8; ++r) {
      float s0 = mk0 ? c[0][r] : -1.0e30f;
      float s1 = mk1 ? c[1][r] : -1.0e30f;
      float mx = half_max(fmaxf(s0, s1));
      float mn = fmaxf(m_[r], mx);
      float ps = half_sum(__expf(s0 - mn) + __expf(s1 - mn));
      l_[r] = l_[r] * __expf(m_[r] - mn) + ps;
      m_[r] = mn;
    }
  }
  float inv_[8];
  #pragma unroll
  for (int r = 0; r < 8; ++r) inv_[r] = 1.0f / l_[r];

  v8f oacc[4];
  #pragma unroll
  for (int ns = 0; ns < 4; ++ns) oacc[ns] = z8;

  // ----------------- pass 2: probs out + P@V -----------------
  for (int k0 = 0; k0 < S_; k0 += 32) {
    __syncthreads();
    { // K tile: async
      const int row = tid >> 2, seg = tid & 3;
      async_copy16(Kb + (size_t)(k0 + row) * HD_ + seg * 16,     &sK[row * 72 + seg * 16]);
      async_copy16(Kb + (size_t)(k0 + row) * HD_ + seg * 16 + 8, &sK[row * 72 + seg * 16 + 8]);
    }
    { // V tile, stored transposed [hd][key] (needs data massage -> VGPR path)
      #pragma unroll
      for (int it = 0; it < 4; ++it) {
        const int task = tid + it * 128;          // 512 tasks: 32 keys x 16 hd-chunks
        const int key = task >> 4, hc = task & 15;
        const unsigned short* vs = Vb + (size_t)(k0 + key) * HD_ + hc * 4;
        sVt[(hc * 4 + 0) * 40 + key] = vs[0];
        sVt[(hc * 4 + 1) * 40 + key] = vs[1];
        sVt[(hc * 4 + 2) * 40 + key] = vs[2];
        sVt[(hc * 4 + 3) * 40 + key] = vs[3];
      }
    }
    if (k0 + 32 < S_) __builtin_prefetch(Vb + (size_t)(k0 + 32 + (tid >> 2)) * HD_, 0, 1);
    async_wait0();
    __syncthreads();

    v8f c[2]; c[0] = z8; c[1] = z8;
    #pragma unroll
    for (int ks = 0; ks < 2; ++ks) {
      const unsigned short* p = &sK[(ks * 16 + l16) * 72];
      v16bf blo = bfcat(*(const v8bf*)(p + g * 8),      *(const v8bf*)(p + 16 + g * 8));
      v16bf bhi = bfcat(*(const v8bf*)(p + 32 + g * 8), *(const v8bf*)(p + 48 + g * 8));
      c[ks] = wmma_bf16(qa0, blo, c[ks]);
      c[ks] = wmma_bf16(qa1, bhi, c[ks]);
    }
    const int mk0 = mrow[k0 + l16];
    const int mk1 = mrow[k0 + 16 + l16];
    unsigned short* Pw = &sP[wave][0];
    #pragma unroll
    for (int ks = 0; ks < 2; ++ks) {
      const int mkv = ks ? mk1 : mk0;
      #pragma unroll
      for (int r = 0; r < 8; ++r) {
        const float s = mkv ? c[ks][r] : -1.0e30f;
        const float p = __expf(s - m_[r]) * inv_[r];
        const int qrow = qt * 64 + wave * 16 + g * 8 + r;
        attb[(size_t)qrow * S_ + k0 + ks * 16 + l16] = p;      // final probability
        Pw[(g * 8 + r) * 40 + ks * 16 + l16] = f2bfu(p);       // stage for P@V
      }
    }
    // P A-operand: row = l16, key chunks {8g..}, {16+8g..}
    const unsigned short* pp = &Pw[l16 * 40];
    v16bf pa = bfcat(*(const v8bf*)(pp + g * 8), *(const v8bf*)(pp + 16 + g * 8));
    #pragma unroll
    for (int ns = 0; ns < 4; ++ns) {  // B = V^T: col = hd = ns*16 + l16, key chunks {8g..}, {16+8g..}
      const unsigned short* vp = &sVt[(ns * 16 + l16) * 40];
      v16bf vb = bfcat(*(const v8bf*)(vp + g * 8), *(const v8bf*)(vp + 16 + g * 8));
      oacc[ns] = wmma_bf16(pa, vb, oacc[ns]);
    }
  }

  // write O in [B*S, D] layout (already transposed back: col = h*64 + hd)
  #pragma unroll
  for (int ns = 0; ns < 4; ++ns) {
    const int col = h * HD_ + ns * 16 + l16;
    #pragma unroll
    for (int r = 0; r < 8; ++r) {
      const int qrow = qt * 64 + wave * 16 + g * 8 + r;
      Obuf[(size_t)(b * S_ + qrow) * D_ + col] = oacc[ns][r];
    }
  }
}

// ---------------------------------------------------------------------------
extern "C" void kernel_launch(void* const* d_in, const int* in_sizes, int n_in,
                              void* d_out, int out_size, void* d_ws, size_t ws_size,
                              hipStream_t stream) {
  (void)in_sizes; (void)n_in; (void)out_size; (void)ws_size;

  const float* query = (const float*)d_in[0];
  const float* key_  = (const float*)d_in[1];
  const float* value = (const float*)d_in[2];
  const int*   mask  = (const int*)  d_in[3];
  const float* Wq = (const float*)d_in[4];
  const float* bq = (const float*)d_in[5];
  const float* Wk = (const float*)d_in[6];
  const float* bk = (const float*)d_in[7];
  const float* Wv = (const float*)d_in[8];
  const float* bv = (const float*)d_in[9];
  const float* Wo = (const float*)d_in[10];
  const float* bo = (const float*)d_in[11];

  const size_t NE = (size_t)B_ * S_ * D_;            // 8,388,608
  unsigned short* Qbf = (unsigned short*)d_ws;
  unsigned short* Kbf = Qbf + NE;
  unsigned short* Vbf = Kbf + NE;
  unsigned short* WtQ = Vbf + NE;
  unsigned short* WtK = WtQ + (size_t)D_ * D_;
  unsigned short* WtV = WtK + (size_t)D_ * D_;
  unsigned short* WtO = WtV + (size_t)D_ * D_;
  float* Obuf = (float*)(WtO + (size_t)D_ * D_);

  float* out_proj = (float*)d_out;                   // [B,S,D]
  float* att_out  = out_proj + NE;                   // [B,H,S,S]

  transpose_cvt_kernel<<<dim3(32, 32, 4), dim3(32, 8), 0, stream>>>(
      Wq, Wk, Wv, Wo, WtQ, WtK, WtV, WtO);

  dim3 ggrid(D_ / 64, (B_ * S_) / 64);               // (16,128)
  gemm_proj_kernel<<<ggrid, 128, 0, stream>>>(query, WtQ, bq, Qbf, nullptr, 0, 0.125f);
  gemm_proj_kernel<<<ggrid, 128, 0, stream>>>(key_,  WtK, bk, Kbf, nullptr, 0, 1.0f);
  gemm_proj_kernel<<<ggrid, 128, 0, stream>>>(value, WtV, bv, Vbf, nullptr, 0, 1.0f);

  attention_kernel<<<dim3(S_ / 64, H_, B_), 128, 0, stream>>>(
      Qbf, Kbf, Vbf, mask, att_out, Obuf);

  gemm_proj_kernel<<<ggrid, 128, 0, stream>>>(Obuf, WtO, bo, nullptr, out_proj, 1, 1.0f);
}
